// LinkFunctionForEvent_592705487033
// MI455X (gfx1250) — matrix operand order repaired and checked by reading, not compiled, not run on hardware
//
#include <hip/hip_runtime.h>
#include <hip/hip_bf16.h>

typedef _Float16     v16h __attribute__((ext_vector_type(16)));
typedef float        v8f  __attribute__((ext_vector_type(8)));
typedef unsigned int v4u  __attribute__((ext_vector_type(4)));

union Frag { v16h h; v4u q[2]; };
union H2   { _Float16 h[2]; unsigned int u; };

// LDS layout (bytes):
//   W1h : [0, 167936)        256 rows * pitch 328 halfs (656B = 41*16B; 164 dw % 64 -> conflict-free frag loads)
//   W2h : [167936, 303104)   256 rows * pitch 264 halfs (528B = 33*16B)
//   Y1b : [303104, 323584)   DOUBLE-BUFFERED: 2 x (128 px * pitch 40 halfs (80B)) f16 Y1 slices
//   b1f : [323584, 324608)   256 f32
//   b2f : [324608, 325632)   256 f32
//   w3f : [325632, 326656)   256 f32
//   outb: [326656, 327168)   128 f32 (layer-3 reduction)
#define SMEM_BYTES 327168
#define Y1_BUF_BYTES 10240

__launch_bounds__(256)
__global__ void edge_mlp_fused(const float* __restrict__ edge,
                               const float* __restrict__ node,
                               const float* __restrict__ W1,
                               const float* __restrict__ b1,
                               const float* __restrict__ W2,
                               const float* __restrict__ b2,
                               const float* __restrict__ W3,
                               const float* __restrict__ b3,
                               float* __restrict__ out)
{
    __shared__ __align__(16) unsigned char smem[SMEM_BYTES];
    _Float16* W1h = (_Float16*)(smem);
    _Float16* W2h = (_Float16*)(smem + 167936);
    unsigned char* Y1base = smem + 303104;
    float* b1f  = (float*)(smem + 323584);
    float* b2f  = (float*)(smem + 324608);
    float* w3f  = (float*)(smem + 325632);
    float* outb = (float*)(smem + 326656);

    const int tid  = threadIdx.x;
    const int wave = tid >> 5;
    const int lane = tid & 31;
    const int lo   = lane & 15;
    const int hi   = lane >> 4;
    const int row2 = wave >> 1;   // layer-2 channel-row of this wave (0..3) -> ct2 in [row2*4, row2*4+4)
    const int col2 = wave & 1;    // layer-2 pixel-column (0..1) -> px in [col2*64, col2*64+64)

    // ---- stage weights f32 -> f16 into LDS (once per persistent block) ----
    for (int idx = tid; idx < 256 * 320; idx += 256) {
        int r = idx / 320, c = idx - r * 320;
        W1h[r * 328 + c] = (_Float16)W1[idx];
    }
    for (int idx = tid; idx < 256 * 256; idx += 256) {
        int r = idx >> 8, c = idx & 255;
        W2h[r * 264 + c] = (_Float16)W2[idx];
    }
    if (tid < 256) { b1f[tid] = b1[tid]; b2f[tid] = b2[tid]; w3f[tid] = W3[tid]; }
    const float b3v = b3[0];
    __syncthreads();

    // 2048 tiles: (b, n, m0) with 128 pixels each; persistent grid-stride loop
    for (int tile = blockIdx.x; tile < 2048; tile += gridDim.x) {
        const int b  = tile >> 9;
        const int r  = tile & 511;
        const int n  = r >> 1;
        const int m0 = (r & 1) << 7;

        __syncthreads();                 // previous tile done with outb / Y1 buffers
        if (tid < 128) outb[tid] = 0.f;

        // ---- build layer-1 B fragments (K x 16px) from global, f32 -> f16 ----
        // B layout (32x16 f16, wave32): lane holds 16 consecutive K at K0 = kt*32 + hi*16, N = lo
        const int m = m0 + wave * 16 + lo;   // this lane's pixel column (wave owns 16 px)
        Frag B1[10];
        #pragma unroll
        for (int kt = 0; kt < 10; ++kt) {
            const int k0 = kt * 32 + hi * 16;
            const float* src; int stride;
            if (k0 < 64)        { src = edge + (((size_t)(b * 64 + k0) * 256 + n) * 256 + m); stride = 65536; }
            else if (k0 < 192)  { src = node + ((size_t)(b * 128 + (k0 - 64)) * 256 + n);     stride = 256;   }
            else                { src = node + ((size_t)(b * 128 + (k0 - 192)) * 256 + m);    stride = 256;   }
            #pragma unroll
            for (int j = 0; j < 16; ++j) B1[kt].h[j] = (_Float16)src[(size_t)j * stride];
        }

        // ---- prefetch next tile's edge data (touched exactly once -> comes from HBM) ----
        {
            const int ntile = tile + gridDim.x;
            if (ntile < 2048) {
                const int nb  = ntile >> 9;
                const int nr  = ntile & 511;
                const int nn  = nr >> 1;
                const int nm  = ((nr & 1) << 7) + wave * 16 + lo;
                #pragma unroll
                for (int kt = 0; kt < 2; ++kt) {       // kt 0..1 covers edge channels k = 0..63
                    const int k0 = kt * 32 + hi * 16;
                    const float* p = edge + (((size_t)(nb * 64 + k0) * 256 + nn) * 256 + nm);
                    #pragma unroll
                    for (int j = 0; j < 16; ++j)
                        __builtin_prefetch(p + (size_t)j * 65536, 0, 1);
                }
            }
        }

        // ---- layer-2 accumulators (64ch x 64px per wave), init with b2 ----
        v8f acc2[16];
        #pragma unroll
        for (int q = 0; q < 4; ++q)
            #pragma unroll
            for (int i = 0; i < 8; ++i) {
                float v = b2f[(row2 * 4 + q) * 16 + hi * 8 + i];
                #pragma unroll
                for (int g = 0; g < 4; ++g) acc2[q * 4 + g][i] = v;
            }

        // ---- fused L1 -> L2: produce 32 Y1 channels per step, consume as L2 K-slice ----
        // Double-buffered staging: one barrier per slice (write buf, barrier, read buf).
        #pragma unroll 2
        for (int cp = 0; cp < 8; ++cp) {
            unsigned char* Y1b = Y1base + (cp & 1) * Y1_BUF_BYTES;
            #pragma unroll
            for (int sub = 0; sub < 2; ++sub) {
                const int ct = cp * 2 + sub;
                v8f acc;
                #pragma unroll
                for (int i = 0; i < 8; ++i) acc[i] = b1f[ct * 16 + hi * 8 + i];
                #pragma unroll
                for (int kt = 0; kt < 10; ++kt) {
                    Frag a;   // A frag: row = ct*16+lo, 16 consecutive K at kt*32 + hi*16
                    const unsigned char* p = (const unsigned char*)W1h + (ct * 16 + lo) * 656 + kt * 64 + hi * 32;
                    a.q[0] = *(const v4u*)p;
                    a.q[1] = *(const v4u*)(p + 16);
                    acc = __builtin_amdgcn_wmma_f32_16x16x32_f16(false, a.h, false, B1[kt].h,
                                                                 (short)0, acc, false, false);
                }
                // relu, cvt f16, stage to Y1b [px][32ch]
                const int px = wave * 16 + lo;
                #pragma unroll
                for (int u = 0; u < 4; ++u) {
                    float x0 = acc[2 * u];     x0 = x0 > 0.f ? x0 : 0.f;
                    float x1 = acc[2 * u + 1]; x1 = x1 > 0.f ? x1 : 0.f;
                    H2 h2; h2.h[0] = (_Float16)x0; h2.h[1] = (_Float16)x1;
                    *(unsigned int*)(Y1b + px * 80 + (sub * 16 + hi * 8 + 2 * u) * 2) = h2.u;
                }
            }
            __syncthreads();             // Y1 slice ready for everyone

            Frag bf[4];                  // L2 B frags: px = col2*64 + g*16 + lo, 16 consecutive ch
            #pragma unroll
            for (int g = 0; g < 4; ++g) {
                const unsigned char* p = Y1b + (col2 * 64 + g * 16 + lo) * 80 + hi * 32;
                bf[g].q[0] = *(const v4u*)p;
                bf[g].q[1] = *(const v4u*)(p + 16);
            }
            #pragma unroll
            for (int q = 0; q < 4; ++q) {
                Frag a;
                const unsigned char* p = (const unsigned char*)W2h + ((row2 * 4 + q) * 16 + lo) * 528 + cp * 64 + hi * 32;
                a.q[0] = *(const v4u*)p;
                a.q[1] = *(const v4u*)(p + 16);
                #pragma unroll
                for (int g = 0; g < 4; ++g)
                    acc2[q * 4 + g] = __builtin_amdgcn_wmma_f32_16x16x32_f16(false, a.h, false, bf[g].h,
                                                                             (short)0, acc2[q * 4 + g], false, false);
            }
        }

        // ---- layer 3: relu + dot with W3, LDS-atomic reduction across waves/half-lanes ----
        float part[4] = {0.f, 0.f, 0.f, 0.f};
        #pragma unroll
        for (int q = 0; q < 4; ++q)
            #pragma unroll
            for (int i = 0; i < 8; ++i) {
                float wv = w3f[(row2 * 4 + q) * 16 + hi * 8 + i];
                #pragma unroll
                for (int g = 0; g < 4; ++g) {
                    float y = acc2[q * 4 + g][i];
                    y = y > 0.f ? y : 0.f;
                    part[g] += wv * y;
                }
            }
        #pragma unroll
        for (int g = 0; g < 4; ++g)
            atomicAdd(&outb[col2 * 64 + g * 16 + lo], part[g]);

        __syncthreads();
        if (tid < 128)
            out[((size_t)(b * 256 + n)) * 256 + m0 + tid] = outb[tid] + b3v;
    }
}

extern "C" void kernel_launch(void* const* d_in, const int* in_sizes, int n_in,
                              void* d_out, int out_size, void* d_ws, size_t ws_size,
                              hipStream_t stream) {
    (void)in_sizes; (void)n_in; (void)d_ws; (void)ws_size; (void)out_size;
    const float* edge = (const float*)d_in[0];
    const float* node = (const float*)d_in[1];
    const float* W1   = (const float*)d_in[2];
    const float* b1   = (const float*)d_in[3];
    const float* W2   = (const float*)d_in[4];
    const float* b2   = (const float*)d_in[5];
    const float* W3   = (const float*)d_in[6];
    const float* b3   = (const float*)d_in[7];
    float* out = (float*)d_out;

    edge_mlp_fused<<<512, 256, 0, stream>>>(edge, node, W1, b1, W2, b2, W3, b3, out);
}
